// GGNNEncoder_8735963480508
// MI455X (gfx1250) — matrix operand order repaired and checked by reading, not compiled, not run on hardware
//
#include <hip/hip_runtime.h>

#define N_NODES 20000
#define IN_DIM  128
#define HID     256
#define N_STEPS 6
#define THREADS 256

typedef __attribute__((ext_vector_type(16))) __bf16 v16bf;
typedef __attribute__((ext_vector_type(8)))  __bf16 v8bf;
typedef __attribute__((ext_vector_type(8)))  float  v8f;

// ---------------- elementwise helpers ----------------
__global__ void f32_to_bf16_kernel(const float* __restrict__ src,
                                   __bf16* __restrict__ dst, int n) {
  int i = blockIdx.x * blockDim.x + threadIdx.x;
  if (i < n) dst[i] = (__bf16)src[i];
}

__global__ void zero_f32_kernel(float* __restrict__ p, int n) {
  int i = blockIdx.x * blockDim.x + threadIdx.x;
  if (i < n) p[i] = 0.0f;
}

// Pack logical B[K x N] (f32) into WMMA-fragment layout, bf16, grouped so one
// output tile's whole K-chain is contiguous:
//   dst[((nt*Ktiles + kt)*32 + lane)*16 + e] = B[kt*32 + (e>>3)*16 + (lane>>4)*8 + (e&7)][nt*16 + (lane&15)]
// transposed=1 means src is stored [N x K] (i.e. we multiply by src^T).
__global__ void pack_b_kernel(const float* __restrict__ src, __bf16* __restrict__ dst,
                              int K, int N, int transposed) {
  int idx = blockIdx.x * blockDim.x + threadIdx.x;
  if (idx >= K * N) return;
  int e      = idx & 15;
  int lane   = (idx >> 4) & 31;
  int tile   = idx >> 9;
  int Ktiles = K >> 5;
  int nt = tile / Ktiles;
  int kt = tile - nt * Ktiles;
  int n = nt * 16 + (lane & 15);
  int k = kt * 32 + (e >> 3) * 16 + (lane >> 4) * 8 + (e & 7);
  float v = transposed ? src[(size_t)n * K + k] : src[(size_t)k * N + n];
  dst[idx] = (__bf16)v;
}

// ---------------- WMMA GEMM ----------------
// C[M x N] = A[M x K](bf16,row-major) * Bpack (+bias) ; optionally bf16 copy.
// KT = K/32. 8 waves/block, one 16-row strip per wave, 2 N-tiles in flight,
// B fragments software-pipelined one k-step ahead of their WMMA consumers.
template<int KT, bool HAS_BIAS, bool WRITE_BF>
__global__ __launch_bounds__(256)
void wmma_gemm_kernel(const __bf16* __restrict__ A,
                      const __bf16* __restrict__ Bp,
                      const float*  __restrict__ bias,
                      float*        __restrict__ C,
                      __bf16*       __restrict__ Cbf,
                      int Mtiles, int Ntiles)
{
  const int wave = threadIdx.x >> 5;
  const int lane = threadIdx.x & 31;
  const int mt = blockIdx.x * (blockDim.x >> 5) + wave;
  if (mt >= Mtiles) return;               // wave-uniform: EXEC stays all-ones
  const int hf = lane >> 4;               // half-wave select
  const int lm = lane & 15;
  const int K = KT * 32;
  const int N = Ntiles * 16;

  // A fragments for the whole K range, kept in registers.
  // ISA 16-bit A 16x32 layout: lane half 0 -> K 0..7 & 16..23 ; half 1 -> K 8..15 & 24..31
  const __bf16* Arow = A + (size_t)(mt * 16 + lm) * K;
  v16bf af[KT];
#pragma unroll
  for (int kt = 0; kt < KT; ++kt) {
    v8bf lo = *(const v8bf*)(Arow + kt * 32 + hf * 8);
    v8bf hi = *(const v8bf*)(Arow + kt * 32 + 16 + hf * 8);
    v16bf a;
#pragma unroll
    for (int i = 0; i < 8; ++i) { a[i] = lo[i]; a[i + 8] = hi[i]; }
    af[kt] = a;
  }

  const v16bf* Bt = (const v16bf*)Bp;     // [nt][KT][32 lanes] of 32B fragments
  for (int nt = 0; nt < Ntiles; nt += 2) {
    if (nt + 2 < Ntiles)                  // hint next tile pair (global_prefetch_b8)
      __builtin_prefetch((const void*)(Bt + ((size_t)nt + 2) * KT * 32 + lane), 0, 3);
    const v16bf* b0p = Bt + ((size_t)nt    ) * KT * 32 + lane;
    const v16bf* b1p = Bt + ((size_t)nt + 1) * KT * 32 + lane;
    v8f c0 = {0.f, 0.f, 0.f, 0.f, 0.f, 0.f, 0.f, 0.f};
    v8f c1 = {0.f, 0.f, 0.f, 0.f, 0.f, 0.f, 0.f, 0.f};
    // software pipeline: B fragments for step kt+1 are issued before the
    // WMMAs of step kt, so waits become partial instead of loadcnt==0.
    v16bf b0 = b0p[0];
    v16bf b1 = b1p[0];
#pragma unroll
    for (int kt = 0; kt < KT; ++kt) {
      v16bf nb0 = b0, nb1 = b1;
      if (kt + 1 < KT) {                  // constant-folded at unroll
        nb0 = b0p[(kt + 1) * 32];
        nb1 = b1p[(kt + 1) * 32];
      }
      c0 = __builtin_amdgcn_wmma_f32_16x16x32_bf16(false, af[kt], false, b0,
                                                   (short)0, c0, false, false);
      c1 = __builtin_amdgcn_wmma_f32_16x16x32_bf16(false, af[kt], false, b1,
                                                   (short)0, c1, false, false);
      b0 = nb0;
      b1 = nb1;
    }
    // C/D layout: lanes 0-15: N=lane, rows mt*16+0..7 ; lanes 16-31: N=lane-16, rows +8..15
    const int n0 = nt * 16 + lm;
    const int n1 = n0 + 16;
    const int r0 = mt * 16 + hf * 8;
    float bv0 = 0.0f, bv1 = 0.0f;
    if (HAS_BIAS) { bv0 = bias[n0]; bv1 = bias[n1]; }
    float* Crow = C + (size_t)r0 * N;
#pragma unroll
    for (int i = 0; i < 8; ++i) {
      float v0 = c0[i] + bv0;
      float v1 = c1[i] + bv1;
      Crow[(size_t)i * N + n0] = v0;
      Crow[(size_t)i * N + n1] = v1;
      if (WRITE_BF) {
        Cbf[(size_t)(r0 + i) * N + n0] = (__bf16)v0;
        Cbf[(size_t)(r0 + i) * N + n1] = (__bf16)v1;
      }
    }
  }
}

// ---------------- edge scatter-add ----------------
// 64 lanes per edge, each lane: one float4 gather + 4 f32 atomic adds.
__global__ void scatter_add_kernel(const float* __restrict__ m,
                                   const long long* __restrict__ ei,
                                   float* __restrict__ agg, int n_edges)
{
  int t = threadIdx.x;
  int e = blockIdx.x * 4 + (t >> 6);
  if (e >= n_edges) return;
  int l64 = t & 63;
  long long src = ei[e];
  long long dst = ei[n_edges + e];
  float4 v = ((const float4*)(m + (size_t)src * HID))[l64];
  float* a = agg + (size_t)dst * HID + l64 * 4;
  atomicAdd(a + 0, v.x);
  atomicAdd(a + 1, v.y);
  atomicAdd(a + 2, v.z);
  atomicAdd(a + 3, v.w);
}

// ---------------- GRU gating ----------------
__global__ void gru_gate_kernel(const float* __restrict__ gi,
                                const float* __restrict__ gh,
                                float* __restrict__ h,
                                __bf16* __restrict__ hbf)
{
  int idx = blockIdx.x * blockDim.x + threadIdx.x;
  if (idx >= N_NODES * HID) return;
  int node = idx >> 8;
  int j    = idx & (HID - 1);
  const float* gir = gi + (size_t)node * (3 * HID);
  const float* ghr = gh + (size_t)node * (3 * HID);
  float i_r = gir[j], i_z = gir[HID + j], i_n = gir[2 * HID + j];
  float h_r = ghr[j], h_z = ghr[HID + j], h_n = ghr[2 * HID + j];
  float r = 1.0f / (1.0f + __expf(-(i_r + h_r)));
  float z = 1.0f / (1.0f + __expf(-(i_z + h_z)));
  float n = tanhf(i_n + r * h_n);
  float hv = h[idx];
  float out = (1.0f - z) * n + z * hv;
  h[idx]   = out;
  hbf[idx] = (__bf16)out;
}

// ---------------- ReLU + LayerNorm ----------------
__global__ __launch_bounds__(256)
void relu_ln_kernel(const float* __restrict__ h,
                    const float* __restrict__ g,
                    const float* __restrict__ b,
                    float* __restrict__ out)
{
  __shared__ float s1[8], s2[8];
  int node = blockIdx.x;
  int j    = threadIdx.x;
  float v = h[(size_t)node * HID + j];
  v = fmaxf(v, 0.0f);
  float sum = v, sq = v * v;
#pragma unroll
  for (int o = 16; o > 0; o >>= 1) {
    sum += __shfl_xor(sum, o, 32);
    sq  += __shfl_xor(sq,  o, 32);
  }
  if ((j & 31) == 0) { s1[j >> 5] = sum; s2[j >> 5] = sq; }
  __syncthreads();
  float tot = 0.f, tot2 = 0.f;
#pragma unroll
  for (int i = 0; i < 8; ++i) { tot += s1[i]; tot2 += s2[i]; }
  float mu  = tot * (1.0f / HID);
  float var = tot2 * (1.0f / HID) - mu * mu;
  float inv = rsqrtf(var + 1e-5f);
  out[(size_t)node * HID + j] = (v - mu) * inv * g[j] + b[j];
}

// ---------------- host orchestration ----------------
extern "C" void kernel_launch(void* const* d_in, const int* in_sizes, int n_in,
                              void* d_out, int out_size, void* d_ws, size_t ws_size,
                              hipStream_t stream)
{
  const float*     x    = (const float*)d_in[0];
  const long long* ei   = (const long long*)d_in[1];
  const float*     w_in = (const float*)d_in[2];
  const float*     b_in = (const float*)d_in[3];
  const float*     ggnn = (const float*)d_in[4];
  const float*     w_ih = (const float*)d_in[5];
  const float*     w_hh = (const float*)d_in[6];
  const float*     b_ih = (const float*)d_in[7];
  const float*     b_hh = (const float*)d_in[8];
  const float*     ln_g = (const float*)d_in[9];
  const float*     ln_b = (const float*)d_in[10];
  const int n_edges = in_sizes[1] / 2;

  const int HE = N_NODES * HID;       // 5,120,000
  const int GE = N_NODES * 3 * HID;   // 15,360,000
  const int XE = N_NODES * IN_DIM;    // 2,560,000

  char* ws = (char*)d_ws;
  size_t off = 0;
  auto alloc = [&](size_t bytes) -> void* {
    void* p = ws + off;
    off += (bytes + 511) & ~(size_t)511;
    return p;
  };
  float*  h       = (float*) alloc((size_t)HE * 4);
  __bf16* h_bf    = (__bf16*)alloc((size_t)HE * 2);
  float*  m       = (float*) alloc((size_t)HE * 4);
  float*  agg     = (float*) alloc((size_t)HE * 4);
  __bf16* agg_bf  = (__bf16*)alloc((size_t)HE * 2);
  float*  gi      = (float*) alloc((size_t)GE * 4);
  float*  gh      = (float*) alloc((size_t)GE * 4);
  __bf16* x_bf    = (__bf16*)alloc((size_t)XE * 2);
  __bf16* win_pk  = (__bf16*)alloc((size_t)IN_DIM * HID * 2);
  __bf16* ggnn_pk = (__bf16*)alloc((size_t)N_STEPS * HID * HID * 2);
  __bf16* wih_pk  = (__bf16*)alloc((size_t)3 * HID * HID * 2);
  __bf16* whh_pk  = (__bf16*)alloc((size_t)3 * HID * HID * 2);
  (void)ws_size; (void)n_in; (void)out_size;

  const int Mtiles = N_NODES / 16;                  // 1250
  const dim3 gemm_grid((Mtiles + 7) / 8);

  // --- pack weights to bf16 WMMA layouts ---
  pack_b_kernel<<<(IN_DIM * HID + THREADS - 1) / THREADS, THREADS, 0, stream>>>(
      w_in, win_pk, IN_DIM, HID, 0);
  for (int s = 0; s < N_STEPS; ++s)
    pack_b_kernel<<<(HID * HID + THREADS - 1) / THREADS, THREADS, 0, stream>>>(
        ggnn + (size_t)s * HID * HID, ggnn_pk + (size_t)s * HID * HID, HID, HID, 0);
  pack_b_kernel<<<(3 * HID * HID + THREADS - 1) / THREADS, THREADS, 0, stream>>>(
      w_ih, wih_pk, HID, 3 * HID, 1);               // gi = agg @ w_ih^T
  pack_b_kernel<<<(3 * HID * HID + THREADS - 1) / THREADS, THREADS, 0, stream>>>(
      w_hh, whh_pk, HID, 3 * HID, 1);               // gh = h @ w_hh^T
  f32_to_bf16_kernel<<<(XE + THREADS - 1) / THREADS, THREADS, 0, stream>>>(x, x_bf, XE);

  // --- input projection: h = x @ w_in + b_in (also emit bf16 copy) ---
  wmma_gemm_kernel<IN_DIM / 32, true, true><<<gemm_grid, THREADS, 0, stream>>>(
      x_bf, win_pk, b_in, h, h_bf, Mtiles, HID / 16);

  // --- GGNN steps ---
  for (int s = 0; s < N_STEPS; ++s) {
    // m = h @ W_s
    wmma_gemm_kernel<HID / 32, false, false><<<gemm_grid, THREADS, 0, stream>>>(
        h_bf, ggnn_pk + (size_t)s * HID * HID, (const float*)nullptr, m,
        (__bf16*)nullptr, Mtiles, HID / 16);
    // agg = segment_sum(m[src], dst)
    zero_f32_kernel<<<(HE + THREADS - 1) / THREADS, THREADS, 0, stream>>>(agg, HE);
    scatter_add_kernel<<<(n_edges + 3) / 4, THREADS, 0, stream>>>(m, ei, agg, n_edges);
    f32_to_bf16_kernel<<<(HE + THREADS - 1) / THREADS, THREADS, 0, stream>>>(agg, agg_bf, HE);
    // gh = h @ w_hh^T + b_hh ; gi = agg @ w_ih^T + b_ih
    wmma_gemm_kernel<HID / 32, true, false><<<gemm_grid, THREADS, 0, stream>>>(
        h_bf, whh_pk, b_hh, gh, (__bf16*)nullptr, Mtiles, 3 * HID / 16);
    wmma_gemm_kernel<HID / 32, true, false><<<gemm_grid, THREADS, 0, stream>>>(
        agg_bf, wih_pk, b_ih, gi, (__bf16*)nullptr, Mtiles, 3 * HID / 16);
    // GRU gate fusion -> h (and bf16 copy for next step's GEMMs)
    gru_gate_kernel<<<(HE + THREADS - 1) / THREADS, THREADS, 0, stream>>>(gi, gh, h, h_bf);
  }

  // --- ReLU + LayerNorm -> out ---
  relu_ln_kernel<<<N_NODES, THREADS, 0, stream>>>(h, ln_g, ln_b, (float*)d_out);
}
